// TESRNN_38087769981366
// MI455X (gfx1250) — matrix operand order repaired and checked by reading, not compiled, not run on hardware
//
#include <hip/hip_runtime.h>
#include <hip/hip_bf16.h>
#include <math.h>

typedef __bf16 bf16;
typedef __attribute__((ext_vector_type(16))) __bf16 v16bf;
typedef __attribute__((ext_vector_type(8)))  __bf16 v8bf;
typedef __attribute__((ext_vector_type(8)))  float  v8f;

#define T_TRAIN 32768
#define INPUT_SZ 128
#define HD 256
#define G4 1024            // 4*HD gate width
#define N_OUT 32593        // (T - 128 + 1) - 48
#define MP 32640           // N_OUT padded to multiple of 128 (covers max Tp=32604)

// ---------- helpers ----------
__device__ __forceinline__ bf16 f2bf(float f) {
  unsigned u = __builtin_bit_cast(unsigned, f);
  unsigned r = u + 0x7FFFu + ((u >> 16) & 1u);
  return __builtin_bit_cast(bf16, (unsigned short)(r >> 16));
}
__device__ __forceinline__ float bf2f(bf16 h) {
  unsigned u = ((unsigned)__builtin_bit_cast(unsigned short, h)) << 16;
  return __builtin_bit_cast(float, u);
}
__device__ __forceinline__ float sigm(float x) { return 1.0f / (1.0f + expf(-x)); }

union AB16 { v16bf v; v8bf h[2]; };

// ---------- small utility kernels ----------
__global__ void k_cvt_bf16(const float* __restrict__ s, bf16* __restrict__ d, int n) {
  int i = blockIdx.x * 256 + threadIdx.x;
  if (i < n) d[i] = f2bf(s[i]);
}
__global__ void k_bias_comb(const float* __restrict__ a, const float* __restrict__ b,
                            float* __restrict__ o, int n) {
  int i = blockIdx.x * 256 + threadIdx.x;
  if (i < n) o[i] = a[i] + b[i];
}
__global__ void k_zero_bf16(bf16* __restrict__ p, int n) {
  int i = blockIdx.x * 256 + threadIdx.x;
  if (i < n) p[i] = f2bf(0.0f);
}
__global__ void k_resid(const bf16* __restrict__ a, const bf16* __restrict__ b,
                        bf16* __restrict__ o, int n) {
  int i = blockIdx.x * 256 + threadIdx.x;
  if (i < n) o[i] = f2bf(bf2f(a[i]) + bf2f(b[i]));
}

// sequential exponential-smoothing level scan (latency-bound, 1 thread)
__global__ void k_levels(const float* __restrict__ train, const int* __restrict__ idxs,
                         const float* __restrict__ a0, const float* __restrict__ taup,
                         float* __restrict__ levs) {
  if (threadIdx.x != 0 || blockIdx.x != 0) return;
  int ix = idxs[0];
  float a  = 1.0f / (1.0f + expf(-a0[ix]));
  float fl = taup[ix] * 1.0f;  // MAXIMUM = 1.0
  float prev = fmaxf(train[0], fl);
  levs[0] = prev;
  for (int t = 1; t < T_TRAIN; ++t) {
    prev = fmaxf(a * train[t] + (1.0f - a) * prev, fl);
    levs[t] = prev;
  }
}

// normalized windows -> bf16 GEMM input (rows >= N_OUT zeroed)
__global__ void k_windows(const float* __restrict__ train, const float* __restrict__ levs,
                          bf16* __restrict__ x0) {
  int id = blockIdx.x * 256 + threadIdx.x;
  if (id >= MP * INPUT_SZ) return;
  int row = id >> 7, col = id & 127;
  float v = 0.0f;
  if (row < N_OUT) v = train[row + col] / levs[row + INPUT_SZ - 1];
  x0[id] = f2bf(v);
}

// act: per (i,k) max over TD=12 future values, normalized by level
__global__ void k_act(const float* __restrict__ train, const float* __restrict__ levs,
                      float* __restrict__ act) {
  int id = blockIdx.x * 256 + threadIdx.x;
  if (id >= N_OUT * 4) return;
  int i = id >> 2, k = id & 3;
  int iv = i + INPUT_SZ - 1;
  float mx = -3.4e38f;
  int base = iv + 1 + k * 12;
  for (int j = 0; j < 12; ++j) mx = fmaxf(mx, train[base + j]);
  act[id] = mx / levs[iv];
}

// final 4-wide linear head: pred[i,k] = dot(nlout[i,:], scW[k,:]) + scb[k]
__global__ void k_sc(const bf16* __restrict__ nlout, const float* __restrict__ scW,
                     const float* __restrict__ scb, float* __restrict__ pred) {
  int id = blockIdx.x * 256 + threadIdx.x;
  if (id >= N_OUT * 4) return;
  int row = id >> 2, k = id & 3;
  const bf16* xr = nlout + (size_t)row * HD;
  const float* wr = scW + k * HD;
  float s = 0.0f;
  for (int j = 0; j < HD; ++j) s += bf2f(xr[j]) * wr[j];
  pred[id] = s + scb[k];
}

// ---------- WMMA GEMM: C[M x N] = A[M x K] @ W[N x K]^T + bias ----------
// A, W bf16 row-major; M = gridDim.x*128, N = gridDim.y*64 (both padded).
// mode 0: f32 out; mode 1: bf16 tanh out.
__global__ __launch_bounds__(256) void k_gemm_wmma(
    const bf16* __restrict__ A, const bf16* __restrict__ W,
    const float* __restrict__ bias, float* __restrict__ Cf,
    bf16* __restrict__ Cb, int N, int K, int mode) {
  __shared__ alignas(32) bf16 lA[128 * 32];
  __shared__ alignas(32) bf16 lB[64 * 32];
  const int tid = threadIdx.x;
  const int wv = tid >> 5, lane = tid & 31;
  const int nlane = lane & 15, kg = lane >> 4;
  const size_t rowA0 = (size_t)blockIdx.x * 128;
  const size_t rowW0 = (size_t)blockIdx.y * 64;

  v8f acc[4];
  #pragma unroll
  for (int nt = 0; nt < 4; ++nt)
    #pragma unroll
    for (int r = 0; r < 8; ++r) acc[nt][r] = 0.0f;

  for (int kk = 0; kk < K; kk += 32) {
    // stage tiles: A 128x32 (16 halves/thread), W 64x32 (8 halves/thread)
    {
      int r = tid >> 1, hh = tid & 1;
      *(v16bf*)&lA[r * 32 + hh * 16] =
          *(const v16bf*)(A + (rowA0 + r) * K + kk + hh * 16);
      int rw = tid >> 2, q = tid & 3;
      *(v8bf*)&lB[rw * 32 + q * 8] =
          *(const v8bf*)(W + (rowW0 + rw) * K + kk + q * 8);
    }
    if (kk + 32 < K)  // warm next A tile (global_prefetch_b8)
      __builtin_prefetch(A + (rowA0 + (tid >> 1)) * K + kk + 32, 0, 1);
    __syncthreads();

    AB16 a;
    int mA = wv * 16 + nlane;
    a.h[0] = *(const v8bf*)&lA[mA * 32 + kg * 16];
    a.h[1] = *(const v8bf*)&lA[mA * 32 + kg * 16 + 8];
    #pragma unroll
    for (int nt = 0; nt < 4; ++nt) {
      v16bf b = *(const v16bf*)&lB[(nt * 16 + nlane) * 32 + kg * 16];
      acc[nt] = __builtin_amdgcn_wmma_f32_16x16x32_bf16(
          false, a.v, false, b, (short)0, acc[nt], false, false);
    }
    __syncthreads();
  }

  #pragma unroll
  for (int nt = 0; nt < 4; ++nt) {
    int n = (int)rowW0 + nt * 16 + nlane;
    float bs = bias[n];
    #pragma unroll
    for (int r = 0; r < 8; ++r) {
      size_t m = rowA0 + wv * 16 + r + 8 * kg;
      float v = acc[nt][r] + bs;
      if (mode == 0) Cf[m * N + n] = v;
      else           Cb[m * N + n] = f2bf(tanhf(v));
    }
  }
}

// ---------- persistent dilated-LSTM recurrence ----------
// One workgroup, 16 waves. Per step: g(16x1024) = xproj[t-rows] + h(16x256)@Whh^T,
// Whh streamed from L2 each step; gates via LDS; c-state in registers.
__global__ __launch_bounds__(512) void k_lstm_rec(
    const float* __restrict__ xproj, const bf16* __restrict__ Whh,
    bf16* __restrict__ hout, int d, int nsteps) {
  __shared__ alignas(32) bf16 hA[16 * HD];     // 8 KB   current h (A matrix)
  __shared__ alignas(32) float gL[12 * G4];    // 48 KB  gate pre-activations
  const int tid = threadIdx.x;
  const int wv = tid >> 5, lane = tid & 31;
  const int nlane = lane & 15, kg = lane >> 4;
  const int colbase = wv * 64 + nlane;         // 16 waves x 64 cols = 1024
  const int jcol = tid & 255;                  // updater: hidden unit
  const int rbase = (tid >> 8) * 6;            // updater: rows {0..5} or {6..11}

  for (int i = tid; i < 16 * HD; i += 512) hA[i] = f2bf(0.0f);
  float cst[6] = {0, 0, 0, 0, 0, 0};
  __syncthreads();

  for (int s = 0; s < nsteps; ++s) {
    // init accumulators with precomputed x-projection (bias already folded in)
    v8f acc[4];
    #pragma unroll
    for (int nt = 0; nt < 4; ++nt) {
      int col = colbase + nt * 16;
      #pragma unroll
      for (int r = 0; r < 8; ++r) {
        int m = r + 8 * kg;
        float v = 0.0f;
        if (m < d) v = xproj[(size_t)(s * d + m) * G4 + col];
        acc[nt][r] = v;
      }
    }
    // h @ Whh^T : 8 K-steps x 4 N-tiles of v_wmma_f32_16x16x32_bf16
    #pragma unroll
    for (int kt = 0; kt < 8; ++kt) {
      AB16 a;
      a.h[0] = *(const v8bf*)&hA[nlane * HD + kt * 32 + kg * 16];
      a.h[1] = *(const v8bf*)&hA[nlane * HD + kt * 32 + kg * 16 + 8];
      #pragma unroll
      for (int nt = 0; nt < 4; ++nt) {
        v16bf b = *(const v16bf*)(Whh + (size_t)(colbase + nt * 16) * HD +
                                  kt * 32 + kg * 16);
        acc[nt] = __builtin_amdgcn_wmma_f32_16x16x32_bf16(
            false, a.v, false, b, (short)0, acc[nt], false, false);
      }
    }
    // spill live gate rows to LDS
    #pragma unroll
    for (int nt = 0; nt < 4; ++nt) {
      int col = colbase + nt * 16;
      #pragma unroll
      for (int r = 0; r < 8; ++r) {
        int m = r + 8 * kg;
        if (m < 12) gL[m * G4 + col] = acc[nt][r];
      }
    }
    __syncthreads();
    // gate nonlinearity + state update (c in registers, h -> LDS + global)
    #pragma unroll
    for (int rr = 0; rr < 6; ++rr) {
      int row = rbase + rr;
      if (row < d) {
        const float* gr = gL + row * G4;
        float gi = gr[jcol], gf = gr[HD + jcol];
        float gc = gr[2 * HD + jcol], go = gr[3 * HD + jcol];
        float c = sigm(gf) * cst[rr] + sigm(gi) * tanhf(gc);
        float h = sigm(go) * tanhf(c);
        cst[rr] = c;
        bf16 hb = f2bf(h);
        hA[row * HD + jcol] = hb;
        long tg = (long)s * d + row;
        if (tg < N_OUT) hout[(size_t)tg * HD + jcol] = hb;
      }
    }
    __syncthreads();
  }
}

// ---------- host ----------
extern "C" void kernel_launch(void* const* d_in, const int* in_sizes, int n_in,
                              void* d_out, int out_size, void* d_ws, size_t ws_size,
                              hipStream_t stream) {
  (void)in_sizes; (void)n_in; (void)out_size; (void)ws_size;
  const float* train = (const float*)d_in[0];
  const int*   idxs  = (const int*)d_in[1];
  const float* a0    = (const float*)d_in[2];
  const float* taup  = (const float*)d_in[3];
  const float *Wih[4], *Whh[4], *bih[4], *bhh[4];
  for (int l = 0; l < 4; ++l) {
    Wih[l] = (const float*)d_in[4 + 4 * l];
    Whh[l] = (const float*)d_in[5 + 4 * l];
    bih[l] = (const float*)d_in[6 + 4 * l];
    bhh[l] = (const float*)d_in[7 + 4 * l];
  }
  const float* nlW = (const float*)d_in[20];
  const float* nlb = (const float*)d_in[21];
  const float* scW = (const float*)d_in[22];
  const float* scb = (const float*)d_in[23];

  char* ws = (char*)d_ws;
  size_t off = 0;
  auto alloc = [&](size_t bytes) -> char* {
    char* p = ws + off;
    off += (bytes + 255) & ~(size_t)255;
    return p;
  };
  float* levs  = (float*)alloc((size_t)T_TRAIN * 4);
  bf16*  x0    = (bf16*)alloc((size_t)MP * INPUT_SZ * 2);
  float* xproj = (float*)alloc((size_t)MP * G4 * 4);
  bf16* hb[4];
  for (int l = 0; l < 4; ++l) hb[l] = (bf16*)alloc((size_t)MP * HD * 2);
  bf16* resid = (bf16*)alloc((size_t)MP * HD * 2);
  bf16* nlout = (bf16*)alloc((size_t)MP * HD * 2);
  bf16* wih_b[4];
  wih_b[0] = (bf16*)alloc((size_t)G4 * INPUT_SZ * 2);
  for (int l = 1; l < 4; ++l) wih_b[l] = (bf16*)alloc((size_t)G4 * HD * 2);
  bf16* whh_b[4];
  for (int l = 0; l < 4; ++l) whh_b[l] = (bf16*)alloc((size_t)G4 * HD * 2);
  bf16* nlW_b = (bf16*)alloc((size_t)HD * HD * 2);
  float* biasL[4];
  for (int l = 0; l < 4; ++l) biasL[l] = (float*)alloc((size_t)G4 * 4);

  auto cvt = [&](const float* s, bf16* d, int n) {
    k_cvt_bf16<<<(n + 255) / 256, 256, 0, stream>>>(s, d, n);
  };
  cvt(Wih[0], wih_b[0], G4 * INPUT_SZ);
  for (int l = 1; l < 4; ++l) cvt(Wih[l], wih_b[l], G4 * HD);
  for (int l = 0; l < 4; ++l) cvt(Whh[l], whh_b[l], G4 * HD);
  cvt(nlW, nlW_b, HD * HD);
  for (int l = 0; l < 4; ++l)
    k_bias_comb<<<4, 256, 0, stream>>>(bih[l], bhh[l], biasL[l], G4);

  k_levels<<<1, 1, 0, stream>>>(train, idxs, a0, taup, levs);
  k_windows<<<(MP * INPUT_SZ) / 256, 256, 0, stream>>>(train, levs, x0);
  int padn = (MP - N_OUT) * HD;
  for (int l = 0; l < 4; ++l)
    k_zero_bf16<<<(padn + 255) / 256, 256, 0, stream>>>(hb[l] + (size_t)N_OUT * HD, padn);

  dim3 blk(256);
  dim3 gProj(MP / 128, G4 / 64);
  const int dil[4]   = {1, 3, 6, 12};
  const int kdim[4]  = {INPUT_SZ, HD, HD, HD};
  const bf16* layerIn[4] = {x0, hb[0], hb[1], hb[2]};
  for (int l = 0; l < 4; ++l) {
    k_gemm_wmma<<<gProj, blk, 0, stream>>>(layerIn[l], wih_b[l], biasL[l],
                                           xproj, nullptr, G4, kdim[l], 0);
    int ns = (N_OUT + dil[l] - 1) / dil[l];
    k_lstm_rec<<<1, 512, 0, stream>>>(xproj, whh_b[l], hb[l], dil[l], ns);
  }

  k_resid<<<(MP * HD) / 256, 256, 0, stream>>>(hb[1], hb[3], resid, MP * HD);
  dim3 gNl(MP / 128, HD / 64);
  k_gemm_wmma<<<gNl, blk, 0, stream>>>(resid, nlW_b, nlb, nullptr, nlout, HD, HD, 1);

  float* pred = (float*)d_out;
  float* actp = pred + (size_t)N_OUT * 4;
  k_sc<<<(N_OUT * 4 + 255) / 256, 256, 0, stream>>>(nlout, scW, scb, pred);
  k_act<<<(N_OUT * 4 + 255) / 256, 256, 0, stream>>>(train, levs, actp);
}
